// LinearAttention_32633161515504
// MI455X (gfx1250) — compile-verified, matrix operand order checked by reading, not run on hardware
//
#include <hip/hip_runtime.h>
#include <hip/hip_bf16.h>

typedef __attribute__((ext_vector_type(16))) _Float16 v16h;
typedef __attribute__((ext_vector_type(8)))  _Float16 v8h;
typedef __attribute__((ext_vector_type(4)))  _Float16 v4h;
typedef __attribute__((ext_vector_type(8)))  float    v8f;

#define S_SEQ 2048
#define D_MOD 1024
#define BATCH 4

// GEMM tiling
#define BM 128
#define BN 128
#define BK 32
#define LDS_ROW 80          // bytes per 32-half row: 64 data + 16 pad (bank-conflict-free)
#define TILE_BYTES (128 * LDS_ROW)   // 10240 per A or B tile
#define STAGE_BYTES (2 * TILE_BYTES) // 20480 per stage (A+B)

// ---------------------------------------------------------------------------
// CDNA5 async copy: 16B global -> LDS, tracked by ASYNCcnt.
// The LDS address operand is derived from the real shared-memory pointer
// (low 32 bits of the flat shared address == LDS byte offset, ISA §10.2),
// so the LDS object escapes and the compiler cannot fold its loads to undef.
// ---------------------------------------------------------------------------
__device__ __forceinline__ void async_copy16(const void* lds_ptr, const _Float16* gaddr) {
  unsigned lds_addr = (unsigned)(uintptr_t)lds_ptr;
  asm volatile("global_load_async_to_lds_b128 %0, %1, off"
               :: "v"(lds_addr), "v"(gaddr) : "memory");
}
__device__ __forceinline__ void wait_async0() {
  asm volatile("s_wait_asynccnt 0" ::: "memory");
}

// ---------------------------------------------------------------------------
// f32 -> f16 elementwise convert, 4 elements/thread
// ---------------------------------------------------------------------------
__global__ __launch_bounds__(256) void cvt_f32_to_f16(const float* __restrict__ in,
                                                      _Float16* __restrict__ out,
                                                      long n4) {
  long i = (long)blockIdx.x * blockDim.x + threadIdx.x;
  if (i >= n4) return;
  float4 f = ((const float4*)in)[i];
  v4h h;
  h[0] = (_Float16)f.x; h[1] = (_Float16)f.y;
  h[2] = (_Float16)f.z; h[3] = (_Float16)f.w;
  ((v4h*)out)[i] = h;
}

// ---------------------------------------------------------------------------
// NT f16 GEMM, f32 accumulate, async-LDS double-buffered:
//   C[m,n] = alpha * sum_k A[m,k]*B[n,k] + bias[n]
// Block: 256 thr = 8 waves (4M x 2N). Block tile 128x128, wave tile 32x64.
// grid = (M/128, N/128, Z).
// ---------------------------------------------------------------------------
__global__ __launch_bounds__(256) void gemm_nt_f16(
    const _Float16* __restrict__ A, long sAz, int lda,
    const _Float16* __restrict__ Bm, long sBz, int ldb,
    const float* __restrict__ bias,
    float alpha,
    float* __restrict__ Cf, long sCfz,
    _Float16* __restrict__ Ch, long sChz,
    int ldc, int K) {
  __shared__ __align__(16) char smem[2 * STAGE_BYTES];  // 40 KB

  const int tid   = threadIdx.x;
  const int lane  = tid & 31;
  const int wave  = tid >> 5;
  const int waveM = wave >> 1;      // 0..3
  const int waveN = wave & 1;       // 0..1
  const long z    = blockIdx.z;
  const int  m_blk = blockIdx.x * BM;
  const int  n_blk = blockIdx.y * BN;

  const _Float16* Ab = A  + z * sAz + (long)m_blk * lda;
  const _Float16* Bb = Bm + z * sBz + (long)n_blk * ldb;

  // per-thread copy assignment: 4 x 16B chunks (2 A rows, 2 B rows)
  const int crow = tid >> 2;        // 0..63
  const int cseg = tid & 3;         // 0..3 -> halves [seg*8, seg*8+8)
  const long aoff0 = (long)crow * lda + cseg * 8;
  const long aoff1 = (long)(crow + 64) * lda + cseg * 8;
  const long boff0 = (long)crow * ldb + cseg * 8;
  const long boff1 = (long)(crow + 64) * ldb + cseg * 8;
  const unsigned l0 = (unsigned)(crow * LDS_ROW + cseg * 16);
  const unsigned l1 = (unsigned)((crow + 64) * LDS_ROW + cseg * 16);

  v8f acc[2][4] = {};

  // prologue: stage k=0 into buffer 0
  async_copy16(smem + l0,              Ab + aoff0);
  async_copy16(smem + l1,              Ab + aoff1);
  async_copy16(smem + TILE_BYTES + l0, Bb + boff0);
  async_copy16(smem + TILE_BYTES + l1, Bb + boff1);
  wait_async0();
  __syncthreads();

  unsigned cur = 0;
  for (int kb = 0; kb < K; kb += BK) {
    const unsigned nxt = cur ^ STAGE_BYTES;
    if (kb + BK < K) {  // stage next tile while computing current
      const int kn = kb + BK;
      async_copy16(smem + nxt + l0,              Ab + aoff0 + kn);
      async_copy16(smem + nxt + l1,              Ab + aoff1 + kn);
      async_copy16(smem + nxt + TILE_BYTES + l0, Bb + boff0 + kn);
      async_copy16(smem + nxt + TILE_BYTES + l1, Bb + boff1 + kn);
    }

    // fragment loads from LDS (ds_load_b128, conflict-free via 80B row pitch)
    const char* Abuf = smem + cur;
    const char* Bbuf = smem + cur + TILE_BYTES;
    const int fl = (lane & 15);
    const int fs = (lane >> 4) << 4;   // byte offset of K-subsegment

    v16h aF[2];
#pragma unroll
    for (int f = 0; f < 2; ++f) {
      const char* p = Abuf + (waveM * 32 + f * 16 + fl) * LDS_ROW + fs;
      v8h lo = *(const v8h*)(p);
      v8h hi = *(const v8h*)(p + 32);
      aF[f] = __builtin_shufflevector(lo, hi, 0,1,2,3,4,5,6,7,8,9,10,11,12,13,14,15);
    }
    v16h bF[4];
#pragma unroll
    for (int t = 0; t < 4; ++t) {
      const char* p = Bbuf + (waveN * 64 + t * 16 + fl) * LDS_ROW + fs;
      v8h lo = *(const v8h*)(p);
      v8h hi = *(const v8h*)(p + 32);
      bF[t] = __builtin_shufflevector(lo, hi, 0,1,2,3,4,5,6,7,8,9,10,11,12,13,14,15);
    }

#pragma unroll
    for (int f = 0; f < 2; ++f)
#pragma unroll
      for (int t = 0; t < 4; ++t)
        acc[f][t] = __builtin_amdgcn_wmma_f32_16x16x32_f16(
            false, aF[f], false, bF[t], (short)0, acc[f][t], false, false);

    wait_async0();      // this wave's copies into nxt have landed
    __syncthreads();    // all waves done reading cur + done copying nxt
    cur = nxt;
  }

  // C/D layout: VGPR r -> row +r (lanes 0-15) / +8+r (lanes 16-31); col = lane&15
  const int colLane = lane & 15;
  const int rowHalf = (lane >> 4) * 8;
#pragma unroll
  for (int f = 0; f < 2; ++f) {
#pragma unroll
    for (int t = 0; t < 4; ++t) {
      const int col = n_blk + waveN * 64 + t * 16 + colLane;
      const float bv = bias ? bias[col] : 0.0f;
#pragma unroll
      for (int r = 0; r < 8; ++r) {
        const long row = m_blk + waveM * 32 + f * 16 + rowHalf + r;
        const float val = acc[f][t][r] * alpha + bv;
        if (Cf) Cf[z * sCfz + row * ldc + col] = val;
        if (Ch) Ch[z * sChz + row * ldc + col] = (_Float16)val;
      }
    }
  }
}

// ---------------------------------------------------------------------------
// Row softmax over D=1024 features, one block per row. f32 in -> f16 out.
// ---------------------------------------------------------------------------
__global__ __launch_bounds__(256) void row_softmax_f16(const float* __restrict__ in,
                                                       _Float16* __restrict__ out) {
  __shared__ float red[256];
  const int tid = threadIdx.x;
  const long row = blockIdx.x;
  const float4 f = ((const float4*)(in + row * D_MOD))[tid];

  float m = fmaxf(fmaxf(f.x, f.y), fmaxf(f.z, f.w));
  red[tid] = m; __syncthreads();
  for (int s = 128; s > 0; s >>= 1) {
    if (tid < s) red[tid] = fmaxf(red[tid], red[tid + s]);
    __syncthreads();
  }
  m = red[0]; __syncthreads();

  const float e0 = __expf(f.x - m), e1 = __expf(f.y - m);
  const float e2 = __expf(f.z - m), e3 = __expf(f.w - m);
  red[tid] = e0 + e1 + e2 + e3; __syncthreads();
  for (int s = 128; s > 0; s >>= 1) {
    if (tid < s) red[tid] += red[tid + s];
    __syncthreads();
  }
  const float inv = 1.0f / red[0];

  v4h h;
  h[0] = (_Float16)(e0 * inv); h[1] = (_Float16)(e1 * inv);
  h[2] = (_Float16)(e2 * inv); h[3] = (_Float16)(e3 * inv);
  ((v4h*)(out + row * D_MOD))[tid] = h;
}

// ---------------------------------------------------------------------------
// Column softmax over S=2048 rows (axis=-2), per batch. f32 in -> f16 out.
// transposeOut: write out[z][col][row] (vp^T for the second GEMM).
// ---------------------------------------------------------------------------
__global__ __launch_bounds__(256) void col_softmax_f16(const float* __restrict__ in,
                                                       _Float16* __restrict__ out,
                                                       int transposeOut) {
  __shared__ float red[256];
  __shared__ float cmax[64];
  __shared__ float csum[64];
  const long z  = blockIdx.z;
  const int col = blockIdx.x * 64 + (threadIdx.x & 63);
  const int grp = threadIdx.x >> 6;
  const float* src = in + z * (long)(S_SEQ * D_MOD);

  float m = -3.4e38f;
  for (int r = grp; r < S_SEQ; r += 4) m = fmaxf(m, src[(long)r * D_MOD + col]);
  red[threadIdx.x] = m; __syncthreads();
  if (threadIdx.x < 64)
    cmax[threadIdx.x] = fmaxf(fmaxf(red[threadIdx.x], red[threadIdx.x + 64]),
                              fmaxf(red[threadIdx.x + 128], red[threadIdx.x + 192]));
  __syncthreads();
  const float cm = cmax[threadIdx.x & 63];

  float s = 0.0f;
  for (int r = grp; r < S_SEQ; r += 4) s += __expf(src[(long)r * D_MOD + col] - cm);
  red[threadIdx.x] = s; __syncthreads();
  if (threadIdx.x < 64)
    csum[threadIdx.x] = red[threadIdx.x] + red[threadIdx.x + 64] +
                        red[threadIdx.x + 128] + red[threadIdx.x + 192];
  __syncthreads();
  const float inv = 1.0f / csum[threadIdx.x & 63];

  if (transposeOut) {
    _Float16* dst = out + z * (long)(D_MOD * S_SEQ) + (long)col * S_SEQ;
    for (int r = grp; r < S_SEQ; r += 4)
      dst[r] = (_Float16)(__expf(src[(long)r * D_MOD + col] - cm) * inv);
  } else {
    _Float16* dst = out + z * (long)(S_SEQ * D_MOD);
    for (int r = grp; r < S_SEQ; r += 4)
      dst[(long)r * D_MOD + col] = (_Float16)(__expf(src[(long)r * D_MOD + col] - cm) * inv);
  }
}

// ---------------------------------------------------------------------------
extern "C" void kernel_launch(void* const* d_in, const int* in_sizes, int n_in,
                              void* d_out, int out_size, void* d_ws, size_t ws_size,
                              hipStream_t stream) {
  const float* q  = (const float*)d_in[0];
  const float* k  = (const float*)d_in[1];
  const float* v  = (const float*)d_in[2];
  const float* Wq = (const float*)d_in[3];
  const float* bq = (const float*)d_in[4];
  const float* Wk = (const float*)d_in[5];
  const float* bk = (const float*)d_in[6];
  const float* Wv = (const float*)d_in[7];
  const float* bv = (const float*)d_in[8];

  const long nX = (long)BATCH * S_SEQ * D_MOD;  // 8,388,608
  const long nW = (long)D_MOD * D_MOD;          // 1,048,576
  const float SCALE = 1.0f / 32.0f;             // 1/sqrt(1024)

  char* ws = (char*)d_ws;
  const size_t szXH = (size_t)nX * 2;  // f16 activations
  const size_t szWH = (size_t)nW * 2;  // f16 weights
  const size_t szR  = (size_t)nX * 4;  // f32 raw projections

  _Float16* xh[3];
  xh[0] = (_Float16*)(ws);
  xh[1] = (_Float16*)(ws + szXH);
  xh[2] = (_Float16*)(ws + 2 * szXH);
  size_t off = 3 * szXH;
  _Float16* wh[3];
  wh[0] = (_Float16*)(ws + off); off += szWH;
  wh[1] = (_Float16*)(ws + off); off += szWH;
  wh[2] = (_Float16*)(ws + off); off += szWH;
  float* raw[3];
  raw[0] = (float*)(ws + off); off += szR;
  raw[1] = (float*)(ws + off); off += szR;
  raw[2] = (float*)(ws + off); off += szR;
  _Float16* qp_h  = (_Float16*)(ws + off); off += szXH;
  _Float16* kp_h  = (_Float16*)(ws + off); off += szXH;
  _Float16* vpT_h = (_Float16*)(ws + off); off += szXH;
  _Float16* attn_h = (_Float16*)ws;  // reuse dead xh region (33.5MB <= 50.3MB)

  float* outP  = (float*)d_out;      // [B,S,D]
  float* attnF = outP + nX;          // [B,S,S]

  const float* xin[3] = {q, k, v};
  const float* win[3] = {Wq, Wk, Wv};
  const float* bin[3] = {bq, bk, bv};

  // 1) convert activations + weights to f16
  for (int i = 0; i < 3; ++i) {
    cvt_f32_to_f16<<<(int)((nX / 4 + 255) / 256), 256, 0, stream>>>(xin[i], xh[i], nX / 4);
    cvt_f32_to_f16<<<(int)((nW / 4 + 255) / 256), 256, 0, stream>>>(win[i], wh[i], nW / 4);
  }

  // 2) projections: raw[i] = x_i @ W_i^T + b_i   (M=8192, N=1024, K=1024)
  for (int i = 0; i < 3; ++i) {
    gemm_nt_f16<<<dim3(BATCH * S_SEQ / BM, D_MOD / BN, 1), 256, 0, stream>>>(
        xh[i], 0, D_MOD, wh[i], 0, D_MOD, bin[i], 1.0f,
        raw[i], 0, (_Float16*)nullptr, 0, D_MOD, D_MOD);
  }

  // 3) softmaxes
  row_softmax_f16<<<BATCH * S_SEQ, 256, 0, stream>>>(raw[0], qp_h);
  col_softmax_f16<<<dim3(D_MOD / 64, 1, BATCH), 256, 0, stream>>>(raw[1], kp_h, 0);
  col_softmax_f16<<<dim3(D_MOD / 64, 1, BATCH), 256, 0, stream>>>(raw[2], vpT_h, 1);

  // 4) attn = SCALE * qp @ kp^T   (per batch: M=2048, N=2048, K=1024)
  gemm_nt_f16<<<dim3(S_SEQ / BM, S_SEQ / BN, BATCH), 256, 0, stream>>>(
      qp_h, (long)S_SEQ * D_MOD, D_MOD,
      kp_h, (long)S_SEQ * D_MOD, D_MOD,
      nullptr, SCALE,
      attnF, (long)S_SEQ * S_SEQ,
      attn_h, (long)S_SEQ * S_SEQ, S_SEQ, D_MOD);

  // 5) output = attn @ (vpT)^T   (per batch: M=2048, N=1024, K=2048)
  gemm_nt_f16<<<dim3(S_SEQ / BM, D_MOD / BN, BATCH), 256, 0, stream>>>(
      attn_h, (long)S_SEQ * S_SEQ, S_SEQ,
      vpT_h, (long)D_MOD * S_SEQ, S_SEQ,
      nullptr, 1.0f,
      outP, (long)S_SEQ * D_MOD,
      (_Float16*)nullptr, 0, D_MOD, S_SEQ);
}
